// RPN_40638980555105
// MI455X (gfx1250) — matrix operand order repaired and checked by reading, not compiled, not run on hardware
//
#include <hip/hip_runtime.h>
#include <math.h>

typedef _Float16 v16h __attribute__((ext_vector_type(16)));
typedef _Float16 v8h  __attribute__((ext_vector_type(8)));
typedef float    v8f  __attribute__((ext_vector_type(8)));

#define N_IMG 8
#define HH 100
#define WW 100
#define PIX_PER_IMG (HH*WW)
#define NPIX (N_IMG*PIX_PER_IMG)     // 80000
#define CIN 256
#define CMID 256
#define KANCH 9
#define NANCH (PIX_PER_IMG*KANCH)    // 90000
#define PRE_NMS 3000
#define POST_NMS 300
#define COLB 47                      // ceil(3000/64)
#define NBINS 4096
#define CAND_CAP 4096
#define KSTEPS 72                    // 2304 / 32

// ---------------------------------------------------------------------------
// Weight prep: w1 (256,256,3,3) fp32 -> Wh[t][m][kk] f16 with
// k = tap*256 + ic ordering, t = k/32, kk = k%32 (tap = ky*3+kx)
// ---------------------------------------------------------------------------
__global__ void prep_w3_kernel(const float* __restrict__ w1, _Float16* __restrict__ Wh) {
    int i = blockIdx.x * 256 + threadIdx.x;
    if (i >= KSTEPS * 256 * 32) return;
    int kk = i & 31;
    int m  = (i >> 5) & 255;
    int t  = i >> 13;
    int tap = t >> 3;
    int ky = tap / 3, kx = tap % 3;
    int ic = ((t & 7) << 5) + kk;
    Wh[i] = (_Float16)w1[((m * CIN + ic) * 3 + ky) * 3 + kx];
}

// wc (18,256) + wr (36,256) -> Wcr (64,256) f16, rows 54..63 zero
__global__ void prep_wcr_kernel(const float* __restrict__ wc, const float* __restrict__ wr,
                                _Float16* __restrict__ Wcr) {
    int i = blockIdx.x * 256 + threadIdx.x;
    if (i >= 64 * 256) return;
    int m = i >> 8, k = i & 255;
    float v = 0.f;
    if (m < 18)      v = wc[m * 256 + k];
    else if (m < 54) v = wr[(m - 18) * 256 + k];
    Wcr[i] = (_Float16)v;
}

// ---------------------------------------------------------------------------
// 3x3 conv as implicit GEMM with WMMA f16->f32.
// Block: 256 thr (8 waves). Tile: 128 out-ch x 128 pixels.
// Each wave: one 16-ch strip x 128 px = 8 accumulators, 8 WMMAs per A frag.
// K loop: tap(9) x ic-step(8); pixel decomposition hoisted out of K loop.
// ---------------------------------------------------------------------------
__global__ __launch_bounds__(256)
void conv3x3_wmma_kernel(const float* __restrict__ x, const _Float16* __restrict__ Wh,
                         const float* __restrict__ b1, _Float16* __restrict__ Hbuf) {
    __shared__ _Float16 Bs[128 * 40];    // 128 pixels x 32 K (+8 pad)
    const int tid  = threadIdx.x;
    const int lane = tid & 31;
    const int wave = tid >> 5;
    const int pixbase = blockIdx.x * 128;
    const int m0 = blockIdx.y * 128;
    const int ln = lane & 15;
    const int lh = lane >> 4;            // 0/1

    // staging coords: each thread owns pixel pl, contiguous K range [16*half, 16*half+16)
    const int pl   = tid & 127;
    const int half = tid >> 7;           // 0/1
    const int p    = pixbase + pl;
    const int n    = p / PIX_PER_IMG;
    const int rem  = p - n * PIX_PER_IMG;
    const int y    = rem / WW;
    const int xg   = rem - y * WW;
    const float* xbase_n = x + (size_t)n * CIN * PIX_PER_IMG;
    _Float16* bsp = Bs + pl * 40 + 16 * half;

    v8f acc[8] = {};

    const _Float16* awbase = Wh + (size_t)(m0 + wave * 16 + ln) * 32 + lh * 8;

    for (int tap = 0; tap < 9; ++tap) {
        const int dy = tap / 3 - 1;
        const int dx = tap % 3 - 1;
        const int yy = y + dy, xq = xg + dx;
        const bool valid = (yy >= 0 && yy < HH && xq >= 0 && xq < WW);
        const float* rowp = xbase_n + ((size_t)(16 * half) * PIX_PER_IMG) + (yy * WW + xq);

        for (int ics = 0; ics < 8; ++ics) {
            const int t = tap * 8 + ics;

            __syncthreads();
            {   // stage: 16 channel values of one pixel -> 2x ds_store_b128
                const float* src = rowp + (size_t)(ics * 32) * PIX_PER_IMG;
                v8h pk0, pk1;
                #pragma unroll
                for (int it = 0; it < 8; ++it)
                    pk0[it] = (_Float16)(valid ? src[(size_t)it * PIX_PER_IMG] : 0.0f);
                #pragma unroll
                for (int it = 0; it < 8; ++it)
                    pk1[it] = (_Float16)(valid ? src[(size_t)(8 + it) * PIX_PER_IMG] : 0.0f);
                *(v8h*)(bsp)     = pk0;
                *(v8h*)(bsp + 8) = pk1;
            }
            __syncthreads();

            // A fragment (16x32 f16, documented VGPR layout)
            union { v16h v; v8h h[2]; } afr;
            const _Float16* ap = awbase + (size_t)t * (256 * 32);
            afr.h[0] = *(const v8h*)(ap);
            afr.h[1] = *(const v8h*)(ap + 16);
            if (t < KSTEPS - 1)
                __builtin_prefetch(ap + 256 * 32, 0, 1);

            // batch all 8 B fragments, then 8 back-to-back WMMAs
            union { v16h v; v8h h[2]; } bfr[8];
            #pragma unroll
            for (int ns = 0; ns < 8; ++ns) {
                const _Float16* bp = &Bs[(ns * 16 + ln) * 40 + lh * 8];
                bfr[ns].h[0] = *(const v8h*)(bp);
                bfr[ns].h[1] = *(const v8h*)(bp + 16);
            }
            #pragma unroll
            for (int ns = 0; ns < 8; ++ns)
                acc[ns] = __builtin_amdgcn_wmma_f32_16x16x32_f16(
                    false, afr.v, false, bfr[ns].v, (short)0, acc[ns], false, false);
        }
    }

    // Epilogue: bias + relu, pack 8 contiguous channels per lane, f16 store
    const int cstart = m0 + wave * 16 + lh * 8;
    float bias[8];
    #pragma unroll
    for (int r = 0; r < 8; ++r) bias[r] = b1[cstart + r];
    #pragma unroll
    for (int ns = 0; ns < 8; ++ns) {
        int pix = pixbase + ns * 16 + ln;
        v8h outv;
        #pragma unroll
        for (int r = 0; r < 8; ++r) {
            float z = acc[ns][r] + bias[r];
            z = z > 0.f ? z : 0.f;
            outv[r] = (_Float16)z;
        }
        *(v8h*)(Hbuf + (size_t)pix * 256 + cstart) = outv;
    }
}

// ---------------------------------------------------------------------------
// Fused 1x1 cls+reg head: GEMM M=64 (18 cls + 36 reg + 10 pad), K=256.
// Block 256 thr (8 waves) = 4 M-subtiles x 2 N-subtiles of 16 px.
// Writes cls logits (N,2,90000), reg (N,90000,4), fg scores S.
// ---------------------------------------------------------------------------
__global__ __launch_bounds__(256)
void head1x1_wmma_kernel(const _Float16* __restrict__ Hbuf, const _Float16* __restrict__ Wcr,
                         const float* __restrict__ bc, const float* __restrict__ br,
                         float* __restrict__ cls_out, float* __restrict__ reg_out,
                         float* __restrict__ S) {
    const int tid  = threadIdx.x;
    const int lane = tid & 31;
    const int wave = tid >> 5;
    const int msub = wave & 3;
    const int nsub = wave >> 2;
    const int ln = lane & 15, lh = lane >> 4;
    const int pix = blockIdx.x * 32 + nsub * 16 + ln;

    v8f acc = {};
    const _Float16* bbase = Hbuf + (size_t)pix * 256;
    const _Float16* abase = Wcr + (msub * 16 + ln) * 256;
    #pragma unroll
    for (int ks = 0; ks < 8; ++ks) {
        union { v16h v; v8h h[2]; } afr, bfr;
        const _Float16* ap = abase + ks * 32 + lh * 8;
        afr.h[0] = *(const v8h*)ap;
        afr.h[1] = *(const v8h*)(ap + 16);
        const _Float16* bp = bbase + ks * 32 + lh * 8;
        bfr.h[0] = *(const v8h*)bp;
        bfr.h[1] = *(const v8h*)(bp + 16);
        acc = __builtin_amdgcn_wmma_f32_16x16x32_f16(
            false, afr.v, false, bfr.v, (short)0, acc, false, false);
    }

    const int n   = pix / PIX_PER_IMG;
    const int rem = pix - n * PIX_PER_IMG;
    const int cbase = msub * 16 + lh * 8;   // even -> softmax pairs stay in-lane
    float z0 = 0.f;
    #pragma unroll
    for (int r = 0; r < 8; ++r) {
        int c = cbase + r;
        if (c < 18) {
            float z = acc[r] + bc[c];
            int a = c >> 1, j = c & 1;
            cls_out[(size_t)(n * 2 + j) * NANCH + rem * KANCH + a] = z;
            if (j == 0) z0 = z;
            else        S[(size_t)n * NANCH + rem * KANCH + a] = 1.0f / (1.0f + __expf(z0 - z));
        } else if (c < 54) {
            float z = acc[r] + br[c - 18];
            reg_out[(size_t)n * (NANCH * 4) + rem * 36 + (c - 18)] = z;
        }
    }
}

// ---------------------------------------------------------------------------
// Anchor gen + box decode + clip + min-size mask (folds into score).
// ---------------------------------------------------------------------------
__global__ void decode_kernel(const float* __restrict__ reg_out, float* __restrict__ S,
                              float* __restrict__ Bx, float* __restrict__ anchors_out,
                              const int* __restrict__ pimw, const int* __restrict__ pimh) {
    int A = blockIdx.x * 256 + threadIdx.x;
    int n = blockIdx.y;
    if (A >= NANCH) return;
    int yx = A / KANCH, a = A - yx * KANCH;
    int y = yx / WW, xg = yx - y * WW;
    const float ratios[3] = {0.5f, 1.0f, 2.0f};
    const float scales[3] = {8.0f, 16.0f, 32.0f};
    float r = ratios[a / 3], s = scales[a % 3];
    float hA = 16.0f * s * sqrtf(r);
    float wA = 16.0f * s * sqrtf(1.0f / r);
    float cy0 = y * 16.0f + 8.0f, cx0 = xg * 16.0f + 8.0f;
    float a0 = cy0 - 0.5f * hA, a1 = cx0 - 0.5f * wA;
    float a2 = cy0 + 0.5f * hA, a3 = cx0 + 0.5f * wA;
    if (n == 0) {
        float* ap = anchors_out + (size_t)A * 4;
        ap[0] = a0; ap[1] = a1; ap[2] = a2; ap[3] = a3;
    }
    float ah = a2 - a0, aw = a3 - a1;
    float ay = a0 + 0.5f * ah, ax = a1 + 0.5f * aw;
    const float* lp = reg_out + ((size_t)n * NANCH + A) * 4;
    float dy = lp[0], dx = lp[1], dh = lp[2], dw = lp[3];
    float py = dy * ah + ay, px = dx * aw + ax;
    float bh = __expf(dh) * ah, bw = __expf(dw) * aw;
    float imh = (float)pimh[0], imw = (float)pimw[0];
    float y1 = fminf(fmaxf(py - 0.5f * bh, 0.f), imh);
    float y2 = fminf(fmaxf(py + 0.5f * bh, 0.f), imh);
    float x1 = fminf(fmaxf(px - 0.5f * bw, 0.f), imw);
    float x2 = fminf(fmaxf(px + 0.5f * bw, 0.f), imw);
    float* bp = Bx + ((size_t)n * NANCH + A) * 4;
    bp[0] = y1; bp[1] = x1; bp[2] = y2; bp[3] = x2;
    bool big = (y2 - y1 >= 16.0f) && (x2 - x1 >= 16.0f);
    float sc = S[(size_t)n * NANCH + A];
    S[(size_t)n * NANCH + A] = big ? sc : -1.0f;
}

// ---------------------------------------------------------------------------
// Top-3000 selection: histogram -> threshold bin -> compact -> LDS bitonic sort
// ---------------------------------------------------------------------------
__device__ __forceinline__ int score_bin(float s) {
    int b = (int)((1.0f - s) * 2048.0f);   // descending: high score -> low bin
    return b < 0 ? 0 : (b > NBINS - 1 ? NBINS - 1 : b);
}

__global__ void zero_kernel(int* hist, int* cnt) {
    int i = blockIdx.x * 256 + threadIdx.x;
    if (i < N_IMG * NBINS) hist[i] = 0;
    if (i < N_IMG) cnt[i] = 0;
}

__global__ void hist_kernel(const float* __restrict__ S, int* __restrict__ hist) {
    int A = blockIdx.x * 256 + threadIdx.x;
    int n = blockIdx.y;
    if (A >= NANCH) return;
    atomicAdd(&hist[n * NBINS + score_bin(S[(size_t)n * NANCH + A])], 1);
}

__global__ void thresh_kernel(const int* __restrict__ hist, int* __restrict__ Tbin) {
    int n = blockIdx.x;
    if (threadIdx.x != 0) return;
    int cum = 0, T = NBINS - 1;
    for (int b = 0; b < NBINS; ++b) {
        cum += hist[n * NBINS + b];
        if (cum >= PRE_NMS) { T = b; break; }
    }
    Tbin[n] = T;
}

__global__ void compact_kernel(const float* __restrict__ S, const int* __restrict__ Tbin,
                               int* __restrict__ cnt, float* __restrict__ candS,
                               int* __restrict__ candI) {
    int A = blockIdx.x * 256 + threadIdx.x;
    int n = blockIdx.y;
    if (A >= NANCH) return;
    float s = S[(size_t)n * NANCH + A];
    if (score_bin(s) <= Tbin[n]) {
        int slot = atomicAdd(&cnt[n], 1);
        if (slot < CAND_CAP) {
            candS[n * CAND_CAP + slot] = s;
            candI[n * CAND_CAP + slot] = A;
        }
    }
}

__global__ __launch_bounds__(1024)
void sort_kernel(const float* __restrict__ candS, const int* __restrict__ candI,
                 const int* __restrict__ cnt, const float* __restrict__ Bx,
                 float* __restrict__ sBox, int* __restrict__ sVal) {
    __shared__ float sk[CAND_CAP];
    __shared__ int   si[CAND_CAP];
    int n = blockIdx.x, tid = threadIdx.x;
    int c = cnt[n]; if (c > CAND_CAP) c = CAND_CAP;
    for (int i = tid; i < CAND_CAP; i += 1024) {
        if (i < c) { sk[i] = candS[n * CAND_CAP + i]; si[i] = candI[n * CAND_CAP + i]; }
        else       { sk[i] = -2.0f;                   si[i] = 0x7fffffff; }
    }
    __syncthreads();
    for (int k = 2; k <= CAND_CAP; k <<= 1) {
        for (int j = k >> 1; j > 0; j >>= 1) {
            for (int i = tid; i < CAND_CAP; i += 1024) {
                int l = i ^ j;
                if (l > i) {
                    bool dirDesc = ((i & k) == 0);
                    float ki = sk[i], kl = sk[l];
                    int   ii = si[i], il = si[l];
                    bool iAfter = (ki < kl) || (ki == kl && ii > il);
                    if (iAfter == dirDesc) { sk[i] = kl; sk[l] = ki; si[i] = il; si[l] = ii; }
                }
            }
            __syncthreads();
        }
    }
    for (int r = tid; r < PRE_NMS; r += 1024) {
        float key = sk[r];
        int idx = si[r];
        float b0 = 0.f, b1v = 0.f, b2 = 0.f, b3 = 0.f;
        if (key > -1.5f) {
            const float* bp = Bx + ((size_t)n * NANCH + idx) * 4;
            b0 = bp[0]; b1v = bp[1]; b2 = bp[2]; b3 = bp[3];
        }
        float* op = sBox + ((size_t)n * PRE_NMS + r) * 4;
        op[0] = b0; op[1] = b1v; op[2] = b2; op[3] = b3;
        sVal[n * PRE_NMS + r] = (key >= 0.0f) ? 1 : 0;
    }
}

// ---------------------------------------------------------------------------
// Bitmask NMS: mask[i][colb] bit j set iff IoU(i, colb*64+j) > 0.7 for j > i
// ---------------------------------------------------------------------------
__global__ __launch_bounds__(64)
void nms_mask_kernel(const float* __restrict__ sBox, unsigned long long* __restrict__ mask) {
    int n = blockIdx.z;
    int colb = blockIdx.x, rowb = blockIdx.y;
    __shared__ float cb[64 * 4];
    int tid = threadIdx.x;
    int col0 = colb * 64;
    int ncols = PRE_NMS - col0; if (ncols > 64) ncols = 64;
    if (tid < ncols) {
        const float* bp = sBox + ((size_t)n * PRE_NMS + col0 + tid) * 4;
        cb[tid * 4 + 0] = bp[0]; cb[tid * 4 + 1] = bp[1];
        cb[tid * 4 + 2] = bp[2]; cb[tid * 4 + 3] = bp[3];
    }
    __syncthreads();
    int i = rowb * 64 + tid;
    if (i >= PRE_NMS) return;
    const float* ip = sBox + ((size_t)n * PRE_NMS + i) * 4;
    float y1 = ip[0], x1 = ip[1], y2 = ip[2], x2 = ip[3];
    float ar = (y2 - y1) * (x2 - x1);
    unsigned long long t = 0ULL;
    for (int jj = 0; jj < ncols; ++jj) {
        int j = col0 + jj;
        if (j <= i) continue;
        float yy1 = fmaxf(y1, cb[jj * 4 + 0]);
        float xx1 = fmaxf(x1, cb[jj * 4 + 1]);
        float yy2 = fminf(y2, cb[jj * 4 + 2]);
        float xx2 = fminf(x2, cb[jj * 4 + 3]);
        float inter = fmaxf(yy2 - yy1, 0.f) * fmaxf(xx2 - xx1, 0.f);
        float arj = (cb[jj * 4 + 2] - cb[jj * 4 + 0]) * (cb[jj * 4 + 3] - cb[jj * 4 + 1]);
        float iou = inter / (ar + arj - inter + 1e-9f);
        if (iou > 0.7f) t |= (1ULL << jj);
    }
    mask[((size_t)n * PRE_NMS + i) * COLB + colb] = t;
}

__global__ void nms_reduce_kernel(const float* __restrict__ sBox, const int* __restrict__ sVal,
                                  const unsigned long long* __restrict__ mask,
                                  float* __restrict__ rois) {
    int n = blockIdx.x;
    if (threadIdx.x != 0) return;
    unsigned long long remv[COLB];
    for (int w = 0; w < COLB; ++w) remv[w] = 0ULL;
    int kept = 0;
    for (int i = 0; i < PRE_NMS && kept < POST_NMS; ++i) {
        int w = i >> 6, b = i & 63;
        if (!((remv[w] >> b) & 1ULL) && sVal[n * PRE_NMS + i]) {
            const float* bp = sBox + ((size_t)n * PRE_NMS + i) * 4;
            float* op = rois + ((size_t)n * POST_NMS + kept) * 4;
            op[0] = bp[0]; op[1] = bp[1]; op[2] = bp[2]; op[3] = bp[3];
            kept++;
            const unsigned long long* mp = mask + ((size_t)n * PRE_NMS + i) * COLB;
            for (int w2 = 0; w2 < COLB; ++w2) remv[w2] |= mp[w2];
        }
    }
    for (; kept < POST_NMS; ++kept) {
        float* op = rois + ((size_t)n * POST_NMS + kept) * 4;
        op[0] = op[1] = op[2] = op[3] = 0.f;
    }
}

__global__ void roi_inds_kernel(float* __restrict__ rinds) {
    int i = blockIdx.x * 256 + threadIdx.x;
    if (i < N_IMG * POST_NMS) rinds[i] = (float)(i / POST_NMS);
}

// ---------------------------------------------------------------------------
extern "C" void kernel_launch(void* const* d_in, const int* in_sizes, int n_in,
                              void* d_out, int out_size, void* d_ws, size_t ws_size,
                              hipStream_t stream) {
    (void)in_sizes; (void)n_in; (void)out_size; (void)ws_size;
    const float* x   = (const float*)d_in[0];
    const float* w1  = (const float*)d_in[1];
    const float* b1  = (const float*)d_in[2];
    const float* wc  = (const float*)d_in[3];
    const float* bc  = (const float*)d_in[4];
    const float* wr  = (const float*)d_in[5];
    const float* br  = (const float*)d_in[6];
    const int*   imw = (const int*)d_in[7];
    const int*   imh = (const int*)d_in[8];

    float* out      = (float*)d_out;
    float* cls_out  = out;                         // (8,2,90000)  = 1,440,000
    float* reg_out  = out + 1440000;               // (8,90000,4)  = 2,880,000
    float* rois     = out + 4320000;               // (2400,4)     = 9,600
    float* rinds    = out + 4329600;               // (2400,)      = 2,400
    float* anch     = out + 4332000;               // (90000,4)    = 360,000

    char* ws = (char*)d_ws;
    size_t cur = 0;
    auto alloc = [&](size_t bytes) {
        void* p = ws + cur;
        cur = (cur + bytes + 255) & ~(size_t)255;
        return p;
    };
    _Float16* Wh   = (_Float16*)alloc((size_t)KSTEPS * 256 * 32 * 2);   // 1.18 MB
    _Float16* Wcr  = (_Float16*)alloc((size_t)64 * 256 * 2);            // 32 KB
    _Float16* Hbuf = (_Float16*)alloc((size_t)NPIX * 256 * 2);          // 41 MB
    float*    S    = (float*)alloc((size_t)N_IMG * NANCH * 4);          // 2.9 MB
    float*    Bx   = (float*)alloc((size_t)N_IMG * NANCH * 4 * 4);      // 11.5 MB
    int*      hist = (int*)alloc((size_t)N_IMG * NBINS * 4);
    int*      Tbin = (int*)alloc((size_t)N_IMG * 4);
    int*      cnt  = (int*)alloc((size_t)N_IMG * 4);
    float*    candS= (float*)alloc((size_t)N_IMG * CAND_CAP * 4);
    int*      candI= (int*)alloc((size_t)N_IMG * CAND_CAP * 4);
    float*    sBox = (float*)alloc((size_t)N_IMG * PRE_NMS * 4 * 4);
    int*      sVal = (int*)alloc((size_t)N_IMG * PRE_NMS * 4);
    unsigned long long* mask = (unsigned long long*)alloc((size_t)N_IMG * PRE_NMS * COLB * 8);

    zero_kernel<<<(N_IMG * NBINS + 255) / 256, 256, 0, stream>>>(hist, cnt);
    prep_w3_kernel<<<(KSTEPS * 256 * 32 + 255) / 256, 256, 0, stream>>>(w1, Wh);
    prep_wcr_kernel<<<(64 * 256 + 255) / 256, 256, 0, stream>>>(wc, wr, Wcr);

    conv3x3_wmma_kernel<<<dim3(NPIX / 128, CMID / 128), 256, 0, stream>>>(x, Wh, b1, Hbuf);
    head1x1_wmma_kernel<<<NPIX / 32, 256, 0, stream>>>(Hbuf, Wcr, bc, br, cls_out, reg_out, S);

    decode_kernel<<<dim3((NANCH + 255) / 256, N_IMG), 256, 0, stream>>>(reg_out, S, Bx, anch, imw, imh);
    hist_kernel<<<dim3((NANCH + 255) / 256, N_IMG), 256, 0, stream>>>(S, hist);
    thresh_kernel<<<N_IMG, 32, 0, stream>>>(hist, Tbin);
    compact_kernel<<<dim3((NANCH + 255) / 256, N_IMG), 256, 0, stream>>>(S, Tbin, cnt, candS, candI);
    sort_kernel<<<N_IMG, 1024, 0, stream>>>(candS, candI, cnt, Bx, sBox, sVal);
    nms_mask_kernel<<<dim3(COLB, COLB, N_IMG), 64, 0, stream>>>(sBox, mask);
    nms_reduce_kernel<<<N_IMG, 32, 0, stream>>>(sBox, sVal, mask, rois);
    roi_inds_kernel<<<(N_IMG * POST_NMS + 255) / 256, 256, 0, stream>>>(rinds);
}
